// AdaptiveAttention_37039797960968
// MI455X (gfx1250) — matrix operand order repaired and checked by reading, not compiled, not run on hardware
//
#include <hip/hip_runtime.h>
#include <hip/hip_bf16.h>

typedef __attribute__((ext_vector_type(16))) _Float16 v16h;
typedef __attribute__((ext_vector_type(8)))  float    v8f;

#define Bb 512
#define Rr 196
#define Ee 2048
#define Hh 512

__device__ __forceinline__ float fast_tanh(float x) {
#if __has_builtin(__builtin_amdgcn_tanhf)
    return __builtin_amdgcn_tanhf(x);
#else
    return tanhf(x);
#endif
}

// ---------------------------------------------------------------------------
// Pre-swizzle an (N x K) row-major f32 weight matrix into f16 WMMA B-fragment
// order: dst[((kt*NT + nt)*32 + lane)*16 + j] = src[n*Ktot + k]
//   n = nt*16 + (lane&15),  k = kt*32 + (j>=8 ? 16 : 0) + (lane>=16 ? 8 : 0) + (j&7)
// Each fragment is then one contiguous v16h (32B) load per lane.
// ---------------------------------------------------------------------------
__global__ __launch_bounds__(256) void prep_bfrag(const float* __restrict__ src,
                                                  _Float16* __restrict__ dst,
                                                  int Ktot, int Ntot, int total) {
    int idx = blockIdx.x * 256 + threadIdx.x;
    if (idx >= total) return;
    int j    = idx & 15;
    int lane = (idx >> 4) & 31;
    int frag = idx >> 9;
    int NT   = Ntot >> 4;
    int kt   = frag / NT;
    int nt   = frag - kt * NT;
    int K    = ((j & 8) ? 16 : 0) + ((lane & 16) ? 8 : 0) + (j & 7);
    int n    = nt * 16 + (lane & 15);
    long k   = (long)kt * 32 + K;
    dst[idx] = (_Float16)src[(long)n * Ktot + k];
}

// ---------------------------------------------------------------------------
// Per-batch GEMVs: U_h = hid@U_w.T + U_b ; g = sigmoid(hid.beta + b) ;
// sentinel = g * (hid@S_w.T + S_b).  One block per batch row.
// ---------------------------------------------------------------------------
__global__ __launch_bounds__(256) void hidden_small(const float* __restrict__ hidden,
                                                    const float* __restrict__ U_w,
                                                    const float* __restrict__ U_b,
                                                    const float* __restrict__ S_w,
                                                    const float* __restrict__ S_b,
                                                    const float* __restrict__ beta_w,
                                                    const float* __restrict__ beta_b,
                                                    float* __restrict__ U_h,
                                                    float* __restrict__ sentinel) {
    __shared__ float hs[Hh];
    __shared__ float red[256];
    __shared__ float g_sh;
    int b = blockIdx.x, t = threadIdx.x;
    hs[t]       = hidden[(long)b * Hh + t];
    hs[t + 256] = hidden[(long)b * Hh + t + 256];
    __syncthreads();
    // gate g
    float p = hs[t] * beta_w[t] + hs[t + 256] * beta_w[t + 256];
    red[t] = p;
    __syncthreads();
    for (int s = 128; s > 0; s >>= 1) {
        if (t < s) red[t] += red[t + s];
        __syncthreads();
    }
    if (t == 0) g_sh = 1.0f / (1.0f + __expf(-(red[0] + beta_b[0])));
    __syncthreads();
    float g = g_sh;
    #pragma unroll
    for (int hh = 0; hh < 2; ++hh) {
        int h = t + hh * 256;
        const float4* ur = (const float4*)(U_w + (long)h * Hh);
        const float4* sr = (const float4*)(S_w + (long)h * Hh);
        float u = 0.f, vg = 0.f;
        #pragma unroll 4
        for (int k = 0; k < Hh / 4; ++k) {
            float4 hv = *(const float4*)&hs[k * 4];
            float4 uw = ur[k];
            float4 sw = sr[k];
            u  += hv.x * uw.x + hv.y * uw.y + hv.z * uw.z + hv.w * uw.w;
            vg += hv.x * sw.x + hv.y * sw.y + hv.z * sw.z + hv.w * sw.w;
        }
        U_h[(long)b * Hh + h]      = u + U_b[h];
        sentinel[(long)b * Hh + h] = g * (vg + S_b[h]);
    }
}

// ---------------------------------------------------------------------------
// Fused attention GEMM: for M = B*R rows, N = H, K = E:
//   acc = img_f16 @ W_w_f16^T ; att = tanh(acc + W_b + U_h[b]) ;
//   e[m] = att . v_w + v_b     (att never written to memory)
// Block = 256 threads = 8 waves (2 M-waves x 4 N-waves), M-tile 32, N = 512.
// ---------------------------------------------------------------------------
__global__ __launch_bounds__(256) void attn_gemm(const float* __restrict__ img,
                                                 const _Float16* __restrict__ BfragW,
                                                 const float* __restrict__ U_h,
                                                 const float* __restrict__ W_b,
                                                 const float* __restrict__ v_w,
                                                 const float* __restrict__ v_b,
                                                 float* __restrict__ e_out) {
    __shared__ float As[32 * 36];  // stride 36: bank-conflict-free, 8B aligned
    __shared__ float e_s[32];
    int tid = threadIdx.x;
    int lane = tid & 31, wid = tid >> 5;
    int wm = wid >> 2, wn = wid & 3;
    int m0 = blockIdx.x * 32;
    if (tid < 32) e_s[tid] = 0.0f;

    v8f acc[8];
    #pragma unroll
    for (int t = 0; t < 8; ++t) acc[t] = v8f{0, 0, 0, 0, 0, 0, 0, 0};

    int aM  = wm * 16 + (lane & 15);
    int kbl = (lane >> 4) * 8;
    int ldr = tid >> 3, ldc = (tid & 7) * 4;

    for (int kt = 0; kt < Ee / 32; ++kt) {
        __syncthreads();
        // stage 32x32 f32 A tile (global_load_b128 -> ds_store_b128)
        float4 f = *(const float4*)(img + (long)(m0 + ldr) * Ee + kt * 32 + ldc);
        *(float4*)&As[ldr * 36 + ldc] = f;
        __syncthreads();
        // build f16 A fragment (ISA 16-bit 16x32 A layout)
        v16h a;
        #pragma unroll
        for (int v = 0; v < 8; ++v) {
            int K = ((v & 4) ? 16 : 0) + kbl + 2 * (v & 3);
            const float* sp = &As[aM * 36 + K];
            a[2 * v]     = (_Float16)sp[0];
            a[2 * v + 1] = (_Float16)sp[1];
        }
        #pragma unroll
        for (int t = 0; t < 8; ++t) {
            int nt = wn * 8 + t;
            const _Float16* bp = BfragW + (((long)kt * 32 + nt) * 32 + lane) * 16;
            v16h bfr = *(const v16h*)bp;
            acc[t] = __builtin_amdgcn_wmma_f32_16x16x32_f16(
                false, a, false, bfr, (short)0, acc[t], false, false);
        }
    }
    __syncthreads();
    // fused epilogue: tanh + dot(v) + cross-lane/ cross-wave reduce
    #pragma unroll
    for (int i = 0; i < 8; ++i) {
        int rowInBlk = wm * 16 + i + ((lane & 16) ? 8 : 0);
        int m = m0 + rowInBlk;
        int b = m / Rr;
        float partial = 0.f;
        #pragma unroll
        for (int t = 0; t < 8; ++t) {
            int h = wn * 128 + t * 16 + (lane & 15);
            float x = acc[t][i] + W_b[h] + U_h[(long)b * Hh + h];
            partial += fast_tanh(x) * v_w[h];
        }
        partial += __shfl_xor(partial, 1);
        partial += __shfl_xor(partial, 2);
        partial += __shfl_xor(partial, 4);
        partial += __shfl_xor(partial, 8);  // sum over 16-lane half-group
        if ((lane & 15) == 0) atomicAdd(&e_s[rowInBlk], partial);
    }
    __syncthreads();
    if (tid < 32) e_out[m0 + tid] = e_s[tid] + v_b[0];
}

// ---------------------------------------------------------------------------
// sent_t = sentinel @ T_w.T + T_b  via WMMA (M=512 b, N=2048 e, K=512 h).
// Same core as attn_gemm; plain bias epilogue.
// ---------------------------------------------------------------------------
__global__ __launch_bounds__(256) void sentt_gemm(const float* __restrict__ A,
                                                  const _Float16* __restrict__ BfragT,
                                                  const float* __restrict__ T_b,
                                                  float* __restrict__ out) {
    __shared__ float As[32 * 36];
    const int NTtot = Ee / 16;  // 128 tiles
    int tid = threadIdx.x;
    int lane = tid & 31, wid = tid >> 5;
    int wm = wid >> 2, wn = wid & 3;
    int m0 = blockIdx.x * 32;
    int ntBase = blockIdx.y * 32;  // 512 columns per block

    v8f acc[8];
    #pragma unroll
    for (int t = 0; t < 8; ++t) acc[t] = v8f{0, 0, 0, 0, 0, 0, 0, 0};

    int aM  = wm * 16 + (lane & 15);
    int kbl = (lane >> 4) * 8;
    int ldr = tid >> 3, ldc = (tid & 7) * 4;

    for (int kt = 0; kt < Hh / 32; ++kt) {
        __syncthreads();
        float4 f = *(const float4*)(A + (long)(m0 + ldr) * Hh + kt * 32 + ldc);
        *(float4*)&As[ldr * 36 + ldc] = f;
        __syncthreads();
        v16h a;
        #pragma unroll
        for (int v = 0; v < 8; ++v) {
            int K = ((v & 4) ? 16 : 0) + kbl + 2 * (v & 3);
            const float* sp = &As[aM * 36 + K];
            a[2 * v]     = (_Float16)sp[0];
            a[2 * v + 1] = (_Float16)sp[1];
        }
        #pragma unroll
        for (int t = 0; t < 8; ++t) {
            int nt = ntBase + wn * 8 + t;
            const _Float16* bp = BfragT + (((long)kt * NTtot + nt) * 32 + lane) * 16;
            v16h bfr = *(const v16h*)bp;
            acc[t] = __builtin_amdgcn_wmma_f32_16x16x32_f16(
                false, a, false, bfr, (short)0, acc[t], false, false);
        }
    }
    #pragma unroll
    for (int t = 0; t < 8; ++t) {
        #pragma unroll
        for (int i = 0; i < 8; ++i) {
            int row = m0 + wm * 16 + i + ((lane & 16) ? 8 : 0);
            int col = (ntBase + wn * 8 + t) * 16 + (lane & 15);
            out[(long)row * Ee + col] = acc[t][i] + T_b[col];
        }
    }
}

// ---------------------------------------------------------------------------
// Softmax over [e, 0] and context = img^T alpha + alpha_s * sent_t.
// One block per batch row; pure bandwidth pass over img (L2/HBM).
// ---------------------------------------------------------------------------
__global__ __launch_bounds__(256) void softmax_context(const float* __restrict__ img,
                                                       const float* __restrict__ e_sc,
                                                       const float* __restrict__ sent_t,
                                                       float* __restrict__ ctx_out,
                                                       float* __restrict__ alpha_out) {
    __shared__ float al[Rr];
    __shared__ float red[256];
    __shared__ float s_mx, s_den;
    int b = blockIdx.x, t = threadIdx.x;
    float ev = (t < Rr) ? e_sc[(long)b * Rr + t] : -3.0e38f;
    red[t] = ev;
    __syncthreads();
    for (int s = 128; s > 0; s >>= 1) {
        if (t < s) red[t] = fmaxf(red[t], red[t + s]);
        __syncthreads();
    }
    if (t == 0) s_mx = fmaxf(red[0], 0.0f);  // sentinel logit is 0
    __syncthreads();
    float mx = s_mx;
    float ex = (t < Rr) ? __expf(ev - mx) : 0.0f;
    red[t] = ex;
    __syncthreads();
    for (int s = 128; s > 0; s >>= 1) {
        if (t < s) red[t] += red[t + s];
        __syncthreads();
    }
    if (t == 0) s_den = red[0] + __expf(-mx);
    __syncthreads();
    float den = s_den;
    float alpha_s = __expf(-mx) / den;
    if (t < Rr) {
        float a = ex / den;
        al[t] = a;
        alpha_out[(long)b * Rr + t] = a;
    }
    __syncthreads();
    const float4* imgb = (const float4*)(img + (long)b * Rr * Ee);
    const float4* stb  = (const float4*)(sent_t + (long)b * Ee);
    float4* ctx        = (float4*)(ctx_out + (long)b * Ee);
    #pragma unroll
    for (int c = 0; c < 2; ++c) {
        int e4 = c * 256 + t;  // float4 column index (row = 512 float4)
        float4 st = stb[e4];
        float4 acc4;
        acc4.x = alpha_s * st.x; acc4.y = alpha_s * st.y;
        acc4.z = alpha_s * st.z; acc4.w = alpha_s * st.w;
        for (int r = 0; r < Rr; ++r) {
            float a = al[r];
            float4 v = imgb[(long)r * (Ee / 4) + e4];
            acc4.x += a * v.x; acc4.y += a * v.y;
            acc4.z += a * v.z; acc4.w += a * v.w;
        }
        ctx[e4] = acc4;
    }
}

// ---------------------------------------------------------------------------
extern "C" void kernel_launch(void* const* d_in, const int* in_sizes, int n_in,
                              void* d_out, int out_size, void* d_ws, size_t ws_size,
                              hipStream_t stream) {
    const float* img     = (const float*)d_in[0];   // (B,R,E)
    const float* hidden  = (const float*)d_in[1];   // (B,H)
    const float* U_w     = (const float*)d_in[2];
    const float* U_b     = (const float*)d_in[3];
    const float* W_w     = (const float*)d_in[4];   // (H,E)
    const float* W_b     = (const float*)d_in[5];
    const float* S_w     = (const float*)d_in[6];
    const float* S_b     = (const float*)d_in[7];
    const float* v_w     = (const float*)d_in[8];
    const float* v_b     = (const float*)d_in[9];
    const float* beta_w  = (const float*)d_in[10];
    const float* beta_b  = (const float*)d_in[11];
    const float* T_w     = (const float*)d_in[12];  // (E,H)
    const float* T_b     = (const float*)d_in[13];

    char* ws = (char*)d_ws;                                   // bytes
    _Float16* BfragW  = (_Float16*)(ws);                      // 2 MiB
    _Float16* BfragT  = (_Float16*)(ws + (2u << 20));         // 2 MiB
    float*    U_h     = (float*)(ws + (4u << 20));            // 1 MiB
    float*    sentinel= (float*)(ws + (5u << 20));            // 1 MiB
    float*    e_sc    = (float*)(ws + (6u << 20));            // 0.5 MiB
    float*    sent_t  = (float*)(ws + (13u << 19));           // 4 MiB (offset 6.5 MiB)

    float* ctx_out   = (float*)d_out;                 // (B,E) first output
    float* alpha_out = (float*)d_out + (long)Bb * Ee; // (B,R) second output

    const int fragElems = 1048576;  // both weight matrices: (K/32)*(N/16)*512
    prep_bfrag<<<fragElems / 256, 256, 0, stream>>>(W_w, BfragW, Ee, Hh, fragElems);
    prep_bfrag<<<fragElems / 256, 256, 0, stream>>>(T_w, BfragT, Hh, Ee, fragElems);

    hidden_small<<<Bb, 256, 0, stream>>>(hidden, U_w, U_b, S_w, S_b,
                                         beta_w, beta_b, U_h, sentinel);

    attn_gemm<<<(Bb * Rr) / 32, 256, 0, stream>>>(img, BfragW, U_h, W_b, v_w, v_b, e_sc);

    sentt_gemm<<<dim3(Bb / 32, Ee / 512), 256, 0, stream>>>(sentinel, BfragT, T_b, sent_t);

    softmax_context<<<Bb, 256, 0, stream>>>(img, e_sc, sent_t, ctx_out, alpha_out);
}
